// TrafficGNN_841813590533
// MI455X (gfx1250) — compile-verified, matrix operand order checked by reading, not run on hardware
//
#include <hip/hip_runtime.h>
#include <hip/hip_bf16.h>

typedef __attribute__((ext_vector_type(16))) _Float16 v16h;
typedef __attribute__((ext_vector_type(8)))  _Float16 v8h;
typedef __attribute__((ext_vector_type(8)))  float    v8f;
typedef __attribute__((ext_vector_type(8)))  float    v8fv;
typedef __attribute__((ext_vector_type(16))) float    v16f;

#define N_FEAT 128
#define HIDDEN 64

// ---------------------------------------------------------------- utilities
__global__ void fill_kernel(float* __restrict__ p, float v, int n) {
    int i = blockIdx.x * blockDim.x + threadIdx.x;
    if (i < n) p[i] = v;
}

__global__ void degree_kernel(const int* __restrict__ dst, float* __restrict__ deg, int nE) {
    int e = blockIdx.x * blockDim.x + threadIdx.x;
    if (e < nE) atomicAdd(&deg[dst[e]], 1.0f);
}

__global__ void rsqrt_kernel(float* __restrict__ deg, int n) {
    int i = blockIdx.x * blockDim.x + threadIdx.x;
    if (i < n) deg[i] = rsqrtf(deg[i]);
}

// ------------------------------------------------- WMMA GEMM: C[M,64] = A[M,K] * W[K,64]
// 4 waves/block, TT row tiles (16 rows each) per wave. W is staged transposed
// ([n][K] halves) in LDS so every B fragment is two contiguous 16B runs
// (2x ds_load_b128); every A fragment is two contiguous 32B f32 runs
// (global b128 loads + packed cvt). f32 accumulate via v_wmma_f32_16x16x32_f16.
// Requires M % 16 == 0.
template<int K, int TT>
__global__ __launch_bounds__(128)
void gemm_wmma_kernel(const float* __restrict__ A, const float* __restrict__ W,
                      float* __restrict__ C, int M) {
    __shared__ _Float16 sWt[HIDDEN * K];   // sWt[n*K + k] = (f16) W[k*64 + n]
    for (int i = threadIdx.x; i < HIDDEN * K; i += blockDim.x) {
        const int n = i & (HIDDEN - 1);    // coalesced global read of W[i]
        const int k = i >> 6;
        sWt[n * K + k] = (_Float16)W[i];
    }
    __syncthreads();

    const int wave = threadIdx.x >> 5;
    const int lane = threadIdx.x & 31;
    const int mrow = lane & 15;        // M within tile (A), N within tile (B/C/D)
    const int hi   = (lane >> 4) & 1;  // K-half select (A/B), +8 rows (C/D)

    const int tile0 = (blockIdx.x * 4 + wave) * TT;

    for (int tt = 0; tt < TT; ++tt) {
        const int tile = tile0 + tt;
        if (tile * 16 >= M) return;

        v8f acc[4] = {v8f{}, v8f{}, v8f{}, v8f{}};
        const float* arow = A + (size_t)(tile * 16 + mrow) * K + hi * 8;

#pragma unroll
        for (int k0 = 0; k0 < K; k0 += 32) {
            // A fragment: halves 0-7 <- arow[k0 .. +7], halves 8-15 <- arow[k0+16 .. +7]
            const v8fv r0 = *(const v8fv*)(arow + k0);
            const v8fv r1 = *(const v8fv*)(arow + k0 + 16);
            const v16f af = __builtin_shufflevector(r0, r1,
                0, 1, 2, 3, 4, 5, 6, 7, 8, 9, 10, 11, 12, 13, 14, 15);
            const v16h a = __builtin_convertvector(af, v16h);

#pragma unroll
            for (int t = 0; t < 4; ++t) {
                const _Float16* bp = &sWt[(t * 16 + mrow) * K + k0 + hi * 8];
                const v8h b0 = *(const v8h*)bp;          // ds_load_b128
                const v8h b1 = *(const v8h*)(bp + 16);   // ds_load_b128
                const v16h b = __builtin_shufflevector(b0, b1,
                    0, 1, 2, 3, 4, 5, 6, 7, 8, 9, 10, 11, 12, 13, 14, 15);
                acc[t] = __builtin_amdgcn_wmma_f32_16x16x32_f16(
                    false, a, false, b, (short)0, acc[t], false, false);
            }
        }

        // C/D layout: VGPR r -> row M = r + 8*hi, col N = lane&15
        float* crow = C + (size_t)(tile * 16 + hi * 8) * HIDDEN + mrow;
#pragma unroll
        for (int t = 0; t < 4; ++t)
#pragma unroll
            for (int r = 0; r < 8; ++r)
                crow[(size_t)r * HIDDEN + t * 16] = acc[t][r];
    }
}

// ------------------------------------------- edge scatter: agg[dst] += h[src] * norm
// One wave32 per edge (src/dst uniform per wave -> scalarized loads); each lane
// handles 2 features via a float2 gather and two f32 L2 atomics.
__global__ void scatter_kernel(const float* __restrict__ h,
                               const int* __restrict__ src, const int* __restrict__ dst,
                               const float* __restrict__ dinv,
                               float* __restrict__ agg, int nE) {
    const long long gid = (long long)blockIdx.x * blockDim.x + threadIdx.x;
    const int e = (int)(gid >> 5);
    if (e >= nE) return;
    const int f = ((int)gid & 31) * 2;

    const int s = src[e];
    const int d = dst[e];
    const float norm = dinv[s] * dinv[d];

    const float2 v = *(const float2*)(h + (size_t)s * HIDDEN + f);
    atomicAdd(&agg[(size_t)d * HIDDEN + f],     v.x * norm);
    atomicAdd(&agg[(size_t)d * HIDDEN + f + 1], v.y * norm);
}

// ------------------------- fused self-loop + bias + ReLU: out = relu(agg + h*dinv^2 + b)
__global__ void fuse_relu_kernel(const float* __restrict__ agg, const float* __restrict__ h,
                                 const float* __restrict__ dinv, const float* __restrict__ b,
                                 float* __restrict__ out, int n) {
    int i = blockIdx.x * blockDim.x + threadIdx.x;
    if (i >= n) return;
    const int node = i >> 6;   // HIDDEN == 64
    const int col  = i & 63;
    const float di = dinv[node];
    const float v  = agg[i] + h[i] * di * di + b[col];
    out[i] = v > 0.0f ? v : 0.0f;
}

// ------------------------------------------------- head: out[M,8] = h[M,64] @ Wh + bh
__global__ void head_kernel(const float* __restrict__ h, const float* __restrict__ Wh,
                            const float* __restrict__ bh, float* __restrict__ out, int n) {
    int i = blockIdx.x * blockDim.x + threadIdx.x;
    if (i >= n) return;
    const int node = i >> 3;
    const int a    = i & 7;
    const float* hr = h + (size_t)node * HIDDEN;
    float sum = bh[a];
#pragma unroll 8
    for (int k = 0; k < HIDDEN; ++k)
        sum += hr[k] * Wh[k * 8 + a];
    out[i] = sum;
}

// ---------------------------------------------------------------------------
extern "C" void kernel_launch(void* const* d_in, const int* in_sizes, int n_in,
                              void* d_out, int out_size, void* d_ws, size_t ws_size,
                              hipStream_t stream) {
    const float* x  = (const float*)d_in[0];
    const int* edge = (const int*)d_in[1];   // JAX x64 disabled -> int32
    const float* W1 = (const float*)d_in[2];
    const float* b1 = (const float*)d_in[3];
    const float* W2 = (const float*)d_in[4];
    const float* b2 = (const float*)d_in[5];
    const float* Wh = (const float*)d_in[6];
    const float* bh = (const float*)d_in[7];
    float* out = (float*)d_out;

    const int M = in_sizes[0] / N_FEAT;   // 100000 (multiple of 16)
    const int E = in_sizes[1] / 2;        // 1600000
    const int* src = edge;
    const int* dst = edge + E;

    char* ws = (char*)d_ws;
    float* dinv = (float*)ws;                         // M floats
    float* bufA = (float*)(ws + (size_t)(1 << 19));   // M*HIDDEN floats
    float* bufB = bufA + (size_t)M * HIDDEN;          // M*HIDDEN floats

    const int NB    = 256;
    const int TT    = 4;                        // row tiles per wave
    const int nMH   = M * HIDDEN;
    const int nTil  = (M + 15) / 16;
    const int gGemm = (nTil + 4 * TT - 1) / (4 * TT);
    const int gMH   = (nMH + NB - 1) / NB;
    const int gM    = (M + NB - 1) / NB;
    const int gE    = (E + NB - 1) / NB;
    const int gScat = (E * 32 + NB - 1) / NB;

    // symmetric normalization: dinv = rsqrt(deg + 1)
    fill_kernel<<<gM, NB, 0, stream>>>(dinv, 1.0f, M);
    degree_kernel<<<gE, NB, 0, stream>>>(dst, dinv, E);
    rsqrt_kernel<<<gM, NB, 0, stream>>>(dinv, M);

    // ---- layer 1: h1 = relu(norm-agg(x @ W1) + b1)
    gemm_wmma_kernel<N_FEAT, TT><<<gGemm, 128, 0, stream>>>(x, W1, bufA, M);
    fill_kernel<<<gMH, NB, 0, stream>>>(bufB, 0.0f, nMH);
    scatter_kernel<<<gScat, NB, 0, stream>>>(bufA, src, dst, dinv, bufB, E);
    fuse_relu_kernel<<<gMH, NB, 0, stream>>>(bufB, bufA, dinv, b1, bufA, nMH);

    // ---- layer 2: h2 = relu(norm-agg(h1 @ W2) + b2)
    gemm_wmma_kernel<HIDDEN, TT><<<gGemm, 128, 0, stream>>>(bufA, W2, bufB, M);
    fill_kernel<<<gMH, NB, 0, stream>>>(bufA, 0.0f, nMH);
    scatter_kernel<<<gScat, NB, 0, stream>>>(bufB, src, dst, dinv, bufA, E);
    fuse_relu_kernel<<<gMH, NB, 0, stream>>>(bufA, bufB, dinv, b2, bufB, nMH);

    // ---- head
    head_kernel<<<(M * 8 + NB - 1) / NB, NB, 0, stream>>>(bufB, Wh, bh, out, M * 8);
}
